// GraphTransformerGreedy_29231547417197
// MI455X (gfx1250) — matrix operand order repaired and checked by reading, not compiled, not run on hardware
//
#include <hip/hip_runtime.h>
#include <hip/hip_bf16.h>
#include <math.h>

// ---------------- problem constants ----------------
#define BATCH 512
#define NN    100          // nodes
#define NP    112          // nodes padded to 7 * 16
#define HD    128          // hidden
#define NHEAD 8
#define DHEAD 16
#define H4    512          // FFN hidden
#define LAYERS 2
#define MAXS  200          // 2*N decode steps
#define NEGV  (-1e9f)

typedef _Float16 h16;
typedef __attribute__((ext_vector_type(16))) _Float16 v16h;
typedef __attribute__((ext_vector_type(8)))  float    v8f;

// ---------------- WMMA helpers (CDNA5 16x16x32 f16 -> f32) ----------------
__device__ __forceinline__ v8f wmma_f16(v16h a, v16h b, v8f c) {
  // 8 args: (neg_a, A, neg_b, B, c_mod, C, reuse_a, reuse_b)
  return __builtin_amdgcn_wmma_f32_16x16x32_f16(false, a, false, b, (short)0, c,
                                                false, false);
}

// A-matrix 16x32 (MxK) gather from row-major [.,ld] at (r0, k0).
// ISA layout: lanes 0-15: M=lane, K groups {kb..kb+7, 16+kb..16+kb+7}, kb = 0/8.
__device__ __forceinline__ v16h load_a16(const h16* s, int ld, int r0, int k0, int lane) {
  const int m  = lane & 15;
  const int kb = (lane >> 4) << 3;
  const h16* p = s + (r0 + m) * ld + k0;
  v16h a;
#pragma unroll
  for (int e = 0; e < 16; ++e) {
    int k = (e < 8) ? (kb + e) : (16 + kb + (e - 8));
    a[e] = p[k];
  }
  return a;
}

// A-matrix with K padded: real K = 16 (attention head dim), K>=16 -> 0
__device__ __forceinline__ v16h load_a16_k16(const h16* s, int ld, int r0, int k0, int lane) {
  const int m  = lane & 15;
  const int kb = (lane >> 4) << 3;
  const h16* p = s + (r0 + m) * ld + k0;
  v16h a;
#pragma unroll
  for (int e = 0; e < 16; ++e) {
    int k = (e < 8) ? (kb + e) : (16 + kb + (e - 8));
    a[e] = (k < 16) ? p[k] : (h16)0.0f;
  }
  return a;
}

// B-matrix 32x16 (KxN) where the operand is stored row-major [N][K]
// (i.e. computing X @ W^T with W = [out][in]). K groups {kb..kb+7, 16+kb..16+kb+7}, kb = 0/8.
__device__ __forceinline__ v16h load_bt(const h16* s, int ld, int n0, int k0, int lane) {
  const int n  = lane & 15;
  const int kb = (lane >> 4) << 3;
  const h16* p = s + (n0 + n) * ld + k0;
  v16h b;
#pragma unroll
  for (int e = 0; e < 16; ++e) b[e] = p[(e < 8) ? (kb + e) : (16 + kb + (e - 8))];
  return b;
}

// B-matrix with real K = 16 (attention): K>=16 -> 0
__device__ __forceinline__ v16h load_bt_k16(const h16* s, int ld, int n0, int k0, int lane) {
  const int n  = lane & 15;
  const int kb = (lane >> 4) << 3;
  const h16* p = s + (n0 + n) * ld + k0;
  v16h b;
#pragma unroll
  for (int e = 0; e < 16; ++e) {
    int k = (e < 8) ? (kb + e) : (16 + kb + (e - 8));
    b[e] = (k < 16) ? p[k] : (h16)0.0f;
  }
  return b;
}

// C/D 16x16 f32 scatter
__device__ __forceinline__ void store_c(float* s, int ld, int r0, int c0, int lane, v8f c) {
  const int n  = lane & 15;
  const int mb = (lane >> 4) << 3;
#pragma unroll
  for (int r = 0; r < 8; ++r) s[(r0 + mb + r) * ld + c0 + n] = c[r];
}

// C/D 16x16 scatter with f32->f16 convert on store
__device__ __forceinline__ void store_c16(h16* s, int ld, int r0, int c0, int lane, v8f c) {
  const int n  = lane & 15;
  const int mb = (lane >> 4) << 3;
#pragma unroll
  for (int r = 0; r < 8; ++r) s[(r0 + mb + r) * ld + c0 + n] = (h16)c[r];
}

// ---------------- weight f32 -> f16 prep ----------------
__global__ void cvt_weights_kernel(const float* wqkv, const float* wo,
                                   const float* w1, const float* w2, const float* pw1,
                                   h16* o_qkv, h16* o_wo, h16* o_w1, h16* o_w2, h16* o_pw1) {
  const int i = blockIdx.x * blockDim.x + threadIdx.x;
  const int stride = gridDim.x * blockDim.x;
  for (int j = i; j < LAYERS * 3 * HD * HD; j += stride) o_qkv[j] = (h16)wqkv[j];
  for (int j = i; j < LAYERS * HD * HD;     j += stride) o_wo[j]  = (h16)wo[j];
  for (int j = i; j < LAYERS * H4 * HD;     j += stride) o_w1[j]  = (h16)w1[j];
  for (int j = i; j < LAYERS * HD * H4;     j += stride) o_w2[j]  = (h16)w2[j];
  for (int j = i; j < HD * 2 * HD;          j += stride) o_pw1[j] = (h16)pw1[j];
}

// ---------------- LDS layout (bytes) ----------------
#define OFF_SXF  0                      // f32 [NP][HD]     57344
#define OFF_SXH  57344                  // f16 [NP][HD]     28672  (aliased: P matrix)
#define OFF_BIG  (57344 + 28672)        // f16 scratch      114688 (q,k,vT | ffn hidden | pin)
#define OFF_SATT (OFF_BIG + 114688)     // f32 [NP][HD]     57344  (scores / pre-LN)
#define OFF_SO16 (OFF_SATT + 57344)     // f16 [NP][HD]     28672  (attn out, f16)
#define OFF_SWO  (OFF_SO16 + 28672)     // f16 [HD][HD]     32768  (async-staged Wo panel)
#define SMEM_TOTAL (OFF_SWO + 32768)    // 319488 B <= 320 KB/WGP

__device__ __forceinline__ void layernorm_rows(const float* src, float* dstf, h16* dsth,
                                               const float* w, const float* bb, int tid) {
  for (int r = tid; r < NP; r += 256) {
    const float* row = src + r * HD;
    float m = 0.f;
    for (int c = 0; c < HD; ++c) m += row[c];
    m *= (1.f / HD);
    float var = 0.f;
    for (int c = 0; c < HD; ++c) { float d = row[c] - m; var += d * d; }
    float is = rsqrtf(var * (1.f / HD) + 1e-5f);
    for (int c = 0; c < HD; ++c) {
      float y = (row[c] - m) * is * w[c] + bb[c];
      dstf[r * HD + c] = y;
      dsth[r * HD + c] = (h16)y;
    }
  }
}

// ---------------- encoder + pointer head: one workgroup per batch ----------------
__global__ void __launch_bounds__(256)
encoder_kernel(const float* __restrict__ nf,
               const float* __restrict__ emb_W, const float* __restrict__ emb_b,
               const h16* __restrict__ wqkv_h, const float* __restrict__ bqkv,
               const h16* __restrict__ wo_h,   const float* __restrict__ bo,
               const float* __restrict__ ln1w, const float* __restrict__ ln1b,
               const h16* __restrict__ w1_h,   const float* __restrict__ b1,
               const h16* __restrict__ w2_h,   const float* __restrict__ b2,
               const float* __restrict__ ln2w, const float* __restrict__ ln2b,
               const h16* __restrict__ pw1_h,  const float* __restrict__ pb1,
               const float* __restrict__ pw2,  const float* __restrict__ pb2,
               float* __restrict__ scores_out) {
  extern __shared__ char smem[];
  float* sxf  = (float*)(smem + OFF_SXF);
  h16*   sxh  = (h16*)  (smem + OFF_SXH);     // also reused as P / h1
  h16*   sbig = (h16*)  (smem + OFF_BIG);
  float* satt = (float*)(smem + OFF_SATT);
  h16*   so16 = (h16*)  (smem + OFF_SO16);
  h16*   swo  = (h16*)  (smem + OFF_SWO);

  const int b    = blockIdx.x;
  const int tid  = threadIdx.x;
  const int lane = tid & 31;
  const int wave = tid >> 5;

  // L2 warm hints for the shared f16 weights (global_prefetch_b8)
  __builtin_prefetch(wqkv_h, 0, 1);
  __builtin_prefetch(w1_h, 0, 1);
  __builtin_prefetch(w2_h, 0, 1);

  // ---- embedding: x = nf @ embW^T + emb_b  (scalar, tiny K=3) ----
  const float* nfb = nf + (size_t)b * NN * 3;
  for (int idx = tid; idx < NP * HD; idx += 256) {
    int r = idx / HD, c = idx % HD;
    float v = 0.f;
    if (r < NN) {
      v = emb_b[c];
#pragma unroll
      for (int f = 0; f < 3; ++f) v += nfb[r * 3 + f] * emb_W[c * 3 + f];
    }
    sxf[idx] = v;
    sxh[idx] = (h16)v;
  }
  __syncthreads();

  for (int l = 0; l < LAYERS; ++l) {
    h16* sq  = sbig;                 // [NP][HD]
    h16* sk  = sbig + NP * HD;       // [NP][HD]
    h16* svT = sbig + 2 * NP * HD;   // [HD][HD]  (V transposed, key-padded)

    // zero vT so padded keys contribute exact zeros in P@V
    for (int idx = tid; idx < HD * HD; idx += 256) svT[idx] = (h16)0.f;
    __syncthreads();

    // ---- QKV: [NP,HD] @ Wqkv^T -> 384 cols, 7x24 tiles, K = 4x32 ----
    const h16*   Wq = wqkv_h + (size_t)l * 3 * HD * HD;
    const float* bq = bqkv + l * 3 * HD;
    for (int t = wave; t < 7 * 24; t += 8) {
      int i = t / 24, j = t % 24;
      v8f c = {};
#pragma unroll
      for (int kk = 0; kk < 4; ++kk) {
        v16h a  = load_a16(sxh, HD, i * 16, kk * 32, lane);
        v16h bm = load_bt(Wq, HD, j * 16, kk * 32, lane);
        c = wmma_f16(a, bm, c);
      }
      int n = lane & 15, mb = (lane >> 4) << 3;
#pragma unroll
      for (int r = 0; r < 8; ++r) {
        int row = i * 16 + mb + r, col = j * 16 + n;
        float v = c[r] + bq[col];
        if (col < HD) {
          sq[row * HD + col] = (h16)(v * 0.25f);            // fold 1/sqrt(DH)
        } else if (col < 2 * HD) {
          sk[row * HD + (col - HD)] = (h16)v;
        } else if (row < NN) {
          svT[(col - 2 * HD) * HD + row] = (h16)v;          // V^T, keys>=NN stay 0
        }
      }
    }

    // ---- kick off async DMA of this layer's Wo panel into LDS; it overlaps
    //      the entire attention phase (GLOBAL_LOAD_ASYNC_TO_LDS_B128, ASYNCcnt) ----
    {
      const unsigned long long gbase =
          (unsigned long long)(const void*)(wo_h + (size_t)l * HD * HD);
      for (int idx = tid; idx < (HD * HD) / 8; idx += 256) {   // 2048 x 16B chunks
        unsigned int goff = (unsigned int)(idx * 16);
        unsigned int loff = (unsigned int)(OFF_SWO + idx * 16);
        asm volatile("global_load_async_to_lds_b128 %0, %1, %2"
                     :: "v"(loff), "v"(goff), "s"(gbase) : "memory");
      }
    }
    __syncthreads();

    // ---- attention, head by head ----
    h16* sp = sxh;                                          // P matrix reuses x-f16
    for (int h = 0; h < NHEAD; ++h) {
      // S = Q_h @ K_h^T : 7x7 tiles, K = DH=16 zero-padded to 32
      for (int t = wave; t < 49; t += 8) {
        int i = t / 7, j = t % 7;
        v16h a  = load_a16_k16(sq, HD, i * 16, h * DHEAD, lane);
        v16h bm = load_bt_k16(sk, HD, j * 16, h * DHEAD, lane);
        v8f c = {};
        c = wmma_f16(a, bm, c);
        store_c(satt, HD, i * 16, j * 16, lane, c);
      }
      __syncthreads();
      // row softmax -> P (f16), padded rows/cols forced to 0
      for (int r = tid; r < NP; r += 256) {
        if (r < NN) {
          const float* row = satt + r * HD;
          float mx = -1e30f;
          for (int c = 0; c < NN; ++c) mx = fmaxf(mx, row[c]);
          float sum = 0.f;
          for (int c = 0; c < NN; ++c) sum += __expf(row[c] - mx);
          float inv = 1.f / sum;
          for (int c = 0; c < HD; ++c)
            sp[r * HD + c] = (c < NN) ? (h16)(__expf(row[c] - mx) * inv) : (h16)0.f;
        } else {
          for (int c = 0; c < HD; ++c) sp[r * HD + c] = (h16)0.f;
        }
      }
      __syncthreads();
      // O_h = P @ V_h : 7 tiles (one 16-col band), K = 128 (padded keys)
      for (int t = wave; t < 7; t += 8) {
        v8f c = {};
#pragma unroll
        for (int kk = 0; kk < 4; ++kk) {
          v16h a  = load_a16(sp, HD, t * 16, kk * 32, lane);
          v16h bm = load_bt(svT, HD, h * DHEAD, kk * 32, lane);
          c = wmma_f16(a, bm, c);
        }
        store_c16(so16, HD, t * 16, h * DHEAD, lane, c);    // direct f16 store
      }
      __syncthreads();
    }

    // ---- fence the async Wo panel copy, then o-projection from LDS ----
    asm volatile("s_wait_asynccnt 0x0" ::: "memory");
    __syncthreads();

    const float* bol = bo + l * HD;
    for (int t = wave; t < 7 * 8; t += 8) {
      int i = t / 8, j = t % 8;
      v8f c = {};
#pragma unroll
      for (int kk = 0; kk < 4; ++kk) {
        v16h a  = load_a16(so16, HD, i * 16, kk * 32, lane);
        v16h bm = load_bt(swo, HD, j * 16, kk * 32, lane);   // B from LDS panel
        c = wmma_f16(a, bm, c);
      }
      int n = lane & 15, mb = (lane >> 4) << 3;
#pragma unroll
      for (int r = 0; r < 8; ++r) {
        int row = i * 16 + mb + r, col = j * 16 + n;
        satt[row * HD + col] = sxf[row * HD + col] + c[r] + bol[col];
      }
    }
    __syncthreads();
    layernorm_rows(satt, sxf, sxh, ln1w + l * HD, ln1b + l * HD, tid);
    __syncthreads();

    // ---- FFN: hidden = relu(x @ W1^T + b1), 7x32 tiles ----
    h16* shid = sbig;                                       // [NP][H4]
    const h16*   W1l = w1_h + (size_t)l * H4 * HD;
    const float* b1l = b1 + l * H4;
    for (int t = wave; t < 7 * 32; t += 8) {
      int i = t / 32, j = t % 32;
      v8f c = {};
#pragma unroll
      for (int kk = 0; kk < 4; ++kk) {
        v16h a  = load_a16(sxh, HD, i * 16, kk * 32, lane);
        v16h bm = load_bt(W1l, HD, j * 16, kk * 32, lane);
        c = wmma_f16(a, bm, c);
      }
      int n = lane & 15, mb = (lane >> 4) << 3;
#pragma unroll
      for (int r = 0; r < 8; ++r) {
        int row = i * 16 + mb + r, col = j * 16 + n;
        shid[row * H4 + col] = (h16)fmaxf(c[r] + b1l[col], 0.f);
      }
    }
    __syncthreads();
    // y = hidden @ W2^T + b2 + residual -> satt, then LN2
    const h16*   W2l = w2_h + (size_t)l * HD * H4;
    const float* b2l = b2 + l * HD;
    for (int t = wave; t < 7 * 8; t += 8) {
      int i = t / 8, j = t % 8;
      v8f c = {};
#pragma unroll
      for (int kk = 0; kk < 16; ++kk) {
        v16h a  = load_a16(shid, H4, i * 16, kk * 32, lane);
        v16h bm = load_bt(W2l, H4, j * 16, kk * 32, lane);
        c = wmma_f16(a, bm, c);
      }
      int n = lane & 15, mb = (lane >> 4) << 3;
#pragma unroll
      for (int r = 0; r < 8; ++r) {
        int row = i * 16 + mb + r, col = j * 16 + n;
        satt[row * HD + col] = sxf[row * HD + col] + c[r] + b2l[col];
      }
    }
    __syncthreads();
    layernorm_rows(satt, sxf, sxh, ln2w + l * HD, ln2b + l * HD, tid);
    __syncthreads();
  }

  // ---- pointer head ----
  // ctx = mean over real rows (stash in satt[0..HD))
  for (int c = tid; c < HD; c += 256) {
    float s = 0.f;
    for (int r = 0; r < NN; ++r) s += sxf[r * HD + c];
    satt[c] = s * (1.f / NN);
  }
  __syncthreads();
  // pin = [x | ctx]  f16 [NP][2H]
  h16* pin = sbig;
  for (int idx = tid; idx < NP * 2 * HD; idx += 256) {
    int r = idx >> 8, c = idx & 255;
    pin[idx] = (c < HD) ? sxh[r * HD + c] : (h16)satt[c - HD];
  }
  __syncthreads();
  // h1 = relu(pin @ pW1^T + pb1) : 7x8 tiles, K = 8x32; dest reuses sxh
  for (int t = wave; t < 7 * 8; t += 8) {
    int i = t / 8, j = t % 8;
    v8f c = {};
#pragma unroll
    for (int kk = 0; kk < 8; ++kk) {
      v16h a  = load_a16(pin, 2 * HD, i * 16, kk * 32, lane);
      v16h bm = load_bt(pw1_h, 2 * HD, j * 16, kk * 32, lane);
      c = wmma_f16(a, bm, c);
    }
    int n = lane & 15, mb = (lane >> 4) << 3;
#pragma unroll
    for (int r = 0; r < 8; ++r) {
      int row = i * 16 + mb + r, col = j * 16 + n;
      sxh[row * HD + col] = (h16)fmaxf(c[r] + pb1[col], 0.f);
    }
  }
  __syncthreads();
  // scores = h1 @ pW2^T + pb2 (single output column -> scalar dot)
  for (int r = tid; r < NN; r += 256) {
    float s = pb2[0];
    for (int k = 0; k < HD; ++k) s += (float)sxh[r * HD + k] * pw2[k];
    scores_out[b * NN + r] = s;
  }
}

// ---------------- greedy decode: one wave32 per batch ----------------
__device__ __forceinline__ bool wave_all32(bool v) {
  int x = v ? 1 : 0;
#pragma unroll
  for (int off = 16; off > 0; off >>= 1) x &= __shfl_xor(x, off, 32);
  return x != 0;
}

__global__ void __launch_bounds__(256)
decode_kernel(const float* __restrict__ scores, const float* __restrict__ demands,
              const float* __restrict__ caps, int* __restrict__ routes,
              float* __restrict__ lps_out) {
  const int lane = threadIdx.x & 31;
  const int b = blockIdx.x * 8 + (threadIdx.x >> 5);
  if (b >= BATCH) return;

  float sc[4], dm[4];
  bool vis[4];
#pragma unroll
  for (int s = 0; s < 4; ++s) {
    int n = lane + 32 * s;
    bool ok = n < NN;
    sc[s]  = ok ? scores[b * NN + n] : NEGV;
    dm[s]  = ok ? demands[b * NN + n] : 0.f;
    vis[s] = !ok;
  }
  const float capacity = caps[b];
  float cap = capacity, lps = 0.f;
  int last = 0, pos = 1;
  bool done = false;

  int* rrow = routes + (size_t)b * (MAXS + 1);
  for (int k = lane; k < MAXS + 1; k += 32) rrow[k] = (k == 0) ? 0 : -1;

  for (int step = 0; step < MAXS; ++step) {
    const bool at_depot = (last == 0);

    bool av = true;
#pragma unroll
    for (int s = 0; s < 4; ++s) {
      int n = lane + 32 * s;
      if (n >= 1 && n < NN && !vis[s]) av = false;
    }
    done = done || (wave_all32(av) && at_depot);

    bool msk[4];
    bool am = true;
#pragma unroll
    for (int s = 0; s < 4; ++s) {
      int n = lane + 32 * s;
      bool m = vis[s] || (dm[s] > cap);
      if (n == 0) m = m || at_depot;
      msk[s] = m;
      if (n < NN && !m) am = false;
    }
    const bool allm = wave_all32(am);
    if (allm && !at_depot && lane == 0) msk[0] = false;   // reopen depot
    done = done || (allm && at_depot);

    // masked argmax (first occurrence) + log-softmax of the selected entry
    float mval[4];
    float bval = NEGV;
    int bidx = NN;
#pragma unroll
    for (int s = 0; s < 4; ++s) {
      int n = lane + 32 * s;
      float v = msk[s] ? NEGV : sc[s];
      mval[s] = v;
      if (n < NN && (v > bval || (v == bval && n < bidx))) { bval = v; bidx = n; }
    }
#pragma unroll
    for (int off = 16; off > 0; off >>= 1) {
      float ov = __shfl_xor(bval, off, 32);
      int   oi = __shfl_xor(bidx, off, 32);
      if (ov > bval || (ov == bval && oi < bidx)) { bval = ov; bidx = oi; }
    }
    float psum = 0.f;
#pragma unroll
    for (int s = 0; s < 4; ++s) {
      int n = lane + 32 * s;
      if (n < NN) psum += __expf(mval[s] - bval);         // bval == row max
    }
#pragma unroll
    for (int off = 16; off > 0; off >>= 1) psum += __shfl_xor(psum, off, 32);
    const float sel = -__logf(psum);                      // lp[act] = max - lse

    if (!done) lps += sel;
    const bool active = !done;
    const int act = bidx;
    const bool upd = active && (act != 0);
    const int slot = act >> 5, sl = act & 31;
    if (upd && lane == sl) vis[slot] = true;
    const float dsel = __shfl(dm[slot], sl, 32);
    if (active && act == 0)      cap = capacity;
    else if (upd)                cap = cap - dsel;
    if (active) {
      if (lane == 0) rrow[pos] = act;
      pos += 1;
      last = act;
    }
  }
  if (lane == 0) lps_out[b] = lps;
}

// ---------------- host launcher ----------------
extern "C" void kernel_launch(void* const* d_in, const int* in_sizes, int n_in,
                              void* d_out, int out_size, void* d_ws, size_t ws_size,
                              hipStream_t stream) {
  (void)in_sizes; (void)n_in; (void)out_size; (void)ws_size;
  const float* nf    = (const float*)d_in[0];
  const float* dem   = (const float*)d_in[1];
  const float* capsv = (const float*)d_in[2];
  const float* embW  = (const float*)d_in[3];
  const float* embB  = (const float*)d_in[4];
  const float* Wqkv  = (const float*)d_in[5];
  const float* bqkv  = (const float*)d_in[6];
  const float* Wo    = (const float*)d_in[7];
  const float* bo    = (const float*)d_in[8];
  const float* ln1w  = (const float*)d_in[9];
  const float* ln1b  = (const float*)d_in[10];
  const float* W1    = (const float*)d_in[11];
  const float* b1    = (const float*)d_in[12];
  const float* W2    = (const float*)d_in[13];
  const float* b2    = (const float*)d_in[14];
  const float* ln2w  = (const float*)d_in[15];
  const float* ln2b  = (const float*)d_in[16];
  const float* pW1   = (const float*)d_in[17];
  const float* pb1   = (const float*)d_in[18];
  const float* pW2   = (const float*)d_in[19];
  const float* pb2   = (const float*)d_in[20];

  // d_ws layout
  char* ws = (char*)d_ws;
  float* scores = (float*)ws;                               // 512*100 f32
  size_t off = (size_t)BATCH * NN * sizeof(float);
  h16* qkv_h = (h16*)(ws + off); off += (size_t)LAYERS * 3 * HD * HD * sizeof(h16);
  h16* wo_h  = (h16*)(ws + off); off += (size_t)LAYERS * HD * HD * sizeof(h16);
  h16* w1_h  = (h16*)(ws + off); off += (size_t)LAYERS * H4 * HD * sizeof(h16);
  h16* w2_h  = (h16*)(ws + off); off += (size_t)LAYERS * HD * H4 * sizeof(h16);
  h16* pw1_h = (h16*)(ws + off); off += (size_t)HD * 2 * HD * sizeof(h16);

  cvt_weights_kernel<<<512, 256, 0, stream>>>(Wqkv, Wo, W1, W2, pW1,
                                              qkv_h, wo_h, w1_h, w2_h, pw1_h);

  (void)hipFuncSetAttribute((const void*)encoder_kernel,
                            hipFuncAttributeMaxDynamicSharedMemorySize, SMEM_TOTAL);
  encoder_kernel<<<BATCH, 256, SMEM_TOTAL, stream>>>(
      nf, embW, embB, qkv_h, bqkv, wo_h, bo, ln1w, ln1b,
      w1_h, b1, w2_h, b2, ln2w, ln2b, pw1_h, pb1, pW2, pb2, scores);

  int* routes = (int*)d_out;                                // [512][201] int32
  float* lps  = (float*)((int*)d_out + (size_t)BATCH * (MAXS + 1));
  decode_kernel<<<BATCH / 8, 256, 0, stream>>>(scores, dem, capsv, routes, lps);
}
